// MultiHeadHierarchicalAttentionBlock_80556406604302
// MI455X (gfx1250) — matrix operand order repaired and checked by reading, not compiled, not run on hardware
//
#include <hip/hip_runtime.h>
#include <math.h>

// ---------------------------------------------------------------------------
// CDNA5 (gfx1250) fused implementation of MultiHeadHierarchicalAttentionBlock.
// All GEMMs lowered to v_wmma_f32_16x16x32_bf16 (wave32, 16x16 tiles, K=32).
// ---------------------------------------------------------------------------

typedef __attribute__((ext_vector_type(16))) __bf16 v16bf;
typedef __attribute__((ext_vector_type(8)))  __bf16 v8bf;
typedef __attribute__((ext_vector_type(8)))  float  v8f;

#define SEQ   4096   // S = H*W
#define CCH   256    // C
#define BB    2      // batch
#define NTOK  8192   // B*S
#define SC    64     // split channels / head dim
#define CF    1024   // FFN hidden

__device__ __forceinline__ __bf16 f2bf(float f) {
  unsigned u = __builtin_bit_cast(unsigned, f);
  u = u + 0x7FFFu + ((u >> 16) & 1u);           // round-to-nearest-even
  unsigned short h = (unsigned short)(u >> 16);
  return __builtin_bit_cast(__bf16, h);
}

__device__ __forceinline__ v16bf cat8(v8bf lo, v8bf hi) {
  return __builtin_shufflevector(lo, hi, 0,1,2,3,4,5,6,7,8,9,10,11,12,13,14,15);
}

__device__ __forceinline__ v8f wmma_bf16(v16bf a, v16bf b, v8f c) {
  // D = A(16x32) * B(32x16) + C, f32 accumulate
  return __builtin_amdgcn_wmma_f32_16x16x32_bf16(false, a, false, b, (short)0, c, false, false);
}

// ---------------------------------------------------------------------------
// K0: weight prep. Folds qlw@qw (scaled by 1/sqrt(rank)) and klw@kw into
// combined [16x64] weights; packs [V | Ql | Kl] into one [96x64] bf16 matrix
// per head; converts ow / f1w / f2w to bf16.
// ---------------------------------------------------------------------------
__global__ void prep_kernel(const float* __restrict__ qw, const float* __restrict__ qb,
                            const float* __restrict__ kw, const float* __restrict__ kb,
                            const float* __restrict__ vw, const float* __restrict__ vb,
                            const float* __restrict__ qlw, const float* __restrict__ klw,
                            const float* __restrict__ ow, const float* __restrict__ f1w,
                            const float* __restrict__ f2w,
                            __bf16* __restrict__ Wcat, float* __restrict__ bcat,
                            __bf16* __restrict__ owb, __bf16* __restrict__ f1wb,
                            __bf16* __restrict__ f2wb)
{
  int tid = blockIdx.x * blockDim.x + threadIdx.x;
  int nth = gridDim.x * blockDim.x;
  // combined projection weights [4][96][64]
  for (int e = tid; e < 4 * 96 * 64; e += nth) {
    int i = e / (96 * 64); int rem = e % (96 * 64); int n = rem / 64; int k = rem % 64;
    float v;
    if (n < 64) {
      v = vw[(i * 64 + n) * 64 + k];
    } else if (n < 80) {
      int r = n - 64; float s = 0.f;
      for (int c = 0; c < 64; ++c) s += qlw[(i * 16 + r) * 64 + c] * qw[(i * 64 + c) * 64 + k];
      v = s * 0.25f;                                   // 1/sqrt(RANK)
    } else {
      int r = n - 80; float s = 0.f;
      for (int c = 0; c < 64; ++c) s += klw[(i * 16 + r) * 64 + c] * kw[(i * 64 + c) * 64 + k];
      v = s;
    }
    Wcat[e] = f2bf(v);
  }
  // combined biases [4][96]
  for (int e = tid; e < 4 * 96; e += nth) {
    int i = e / 96; int n = e % 96; float v;
    if (n < 64) v = vb[i * 64 + n];
    else if (n < 80) {
      int r = n - 64; float s = 0.f;
      for (int c = 0; c < 64; ++c) s += qlw[(i * 16 + r) * 64 + c] * qb[i * 64 + c];
      v = s * 0.25f;
    } else {
      int r = n - 80; float s = 0.f;
      for (int c = 0; c < 64; ++c) s += klw[(i * 16 + r) * 64 + c] * kb[i * 64 + c];
      v = s;
    }
    bcat[e] = v;
  }
  for (int e = tid; e < 4 * 64 * 64; e += nth) owb[e]  = f2bf(ow[e]);
  for (int e = tid; e < CF * CCH;    e += nth) f1wb[e] = f2bf(f1w[e]);
  for (int e = tid; e < CCH * CF;    e += nth) f2wb[e] = f2bf(f2w[e]);
}

// ---------------------------------------------------------------------------
// K1: per-head fused projection.  xi = x_slice + prev  (computed on the fly),
// proj = xi @ Wcat^T + bcat  -> V (transposed [B][64][S] bf16), Ql, Kl
// (row-major [NTOK][32] bf16, rank padded 16->32 with zeros).
// One wave per 16-row tile; 6 N-tiles, 2 K-tiles -> 12 WMMAs per wave.
// ---------------------------------------------------------------------------
__global__ void proj_kernel(const float* __restrict__ x, const float* __restrict__ prev,
                            const __bf16* __restrict__ Wcat, const float* __restrict__ bcat,
                            int head,
                            __bf16* __restrict__ Vt, __bf16* __restrict__ Qlb,
                            __bf16* __restrict__ Klb)
{
  int wave = (blockIdx.x * blockDim.x + threadIdx.x) >> 5;   // [0, 512)
  int lane = threadIdx.x & 31, li = lane & 15, half = lane >> 4;
  int mBase = wave * 16;
  int b = mBase >> 12;
  int sBase = mBase & (SEQ - 1);
  int m = mBase + li;                                        // A-row for this lane
  int s = sBase + li;

  const __bf16* Wh = Wcat + head * 96 * 64;
  const float*  bh = bcat + head * 96;
  const float*  xb = x + ((size_t)b * CCH + head * SC) * SEQ + s;   // + k*SEQ
  const float*  pv = prev ? (prev + (size_t)m * SC) : nullptr;

  v16bf afr[2];
#pragma unroll
  for (int t = 0; t < 2; ++t) {
#pragma unroll
    for (int j = 0; j < 16; ++j) {
      int kk = t * 32 + j + half * 16;   // A 16x32 layout
      float v = xb[(size_t)kk * SEQ];
      if (pv) v += pv[kk];
      afr[t][j] = f2bf(v);
    }
  }

  for (int nt = 0; nt < 6; ++nt) {
    int nB = nt * 16;
    v16bf b0 = *(const v16bf*)(Wh + (size_t)(nB + li) * 64 + half * 16);
    v16bf b1 = *(const v16bf*)(Wh + (size_t)(nB + li) * 64 + 32 + half * 16);
    v8f acc{};
    acc = wmma_bf16(afr[0], b0, acc);
    acc = wmma_bf16(afr[1], b1, acc);
    float bias = bh[nB + li];
#pragma unroll
    for (int r = 0; r < 8; ++r) acc[r] += bias;

    if (nt < 4) {                 // V -> transposed store, contiguous in s
      int n = nB + li;
      int s0 = sBase + half * 8;
      v8bf pk;
#pragma unroll
      for (int r = 0; r < 8; ++r) pk[r] = f2bf(acc[r]);
      *(v8bf*)(Vt + ((size_t)(b * SC + n)) * SEQ + s0) = pk;
    } else {                      // Ql / Kl row-major [NTOK][32], pad hi 16 = 0
      __bf16* dst = (nt == 4) ? Qlb : Klb;
#pragma unroll
      for (int r = 0; r < 8; ++r) {
        size_t row = (size_t)(mBase + r + half * 8) * 32;
        dst[row + li]      = f2bf(acc[r]);
        dst[row + 16 + li] = f2bf(0.f);
      }
    }
  }
}

// ---------------------------------------------------------------------------
// K2: flash attention.  One wave owns a 16-query tile; loops over 128 blocks
// of 32 keys.  Per block: 2 score WMMAs, online softmax (shuffle reductions
// within 16-lane halves), P transposed D->A layout via per-wave LDS tile,
// 4 WMMAs accumulating O[16x64].
// ---------------------------------------------------------------------------
__global__ void attn_kernel(const __bf16* __restrict__ Qlb, const __bf16* __restrict__ Klb,
                            const __bf16* __restrict__ Vt, __bf16* __restrict__ Obf)
{
  __shared__ __bf16 ldsP[8][16 * 32];
  int widx = threadIdx.x >> 5;
  int wave = blockIdx.x * 8 + widx;                          // [0, 512)
  int lane = threadIdx.x & 31, li = lane & 15, half = lane >> 4;
  int mBase = wave * 16;
  int b = mBase >> 12;
  int c0 = half * 16;

  int row = mBase + li;
  v8bf qlo = *(const v8bf*)(Qlb + (size_t)row * 32 + c0);
  v8bf qhi = *(const v8bf*)(Qlb + (size_t)row * 32 + c0 + 8);
  v16bf qfr = cat8(qlo, qhi);

  float mrun[8], lrun[8];
#pragma unroll
  for (int r = 0; r < 8; ++r) { mrun[r] = -3.0e38f; lrun[r] = 0.f; }
  v8f accO[4];
#pragma unroll
  for (int n = 0; n < 4; ++n) accO[n] = v8f{};

  const size_t kRowBase = (size_t)b * SEQ;
  __bf16* myP = &ldsP[widx][0];

  for (int kb = 0; kb < SEQ / 32; ++kb) {
    int kBase = kb * 32;
    v16bf kf0 = *(const v16bf*)(Klb + (kRowBase + kBase + li) * 32 + half * 16);
    v16bf kf1 = *(const v16bf*)(Klb + (kRowBase + kBase + 16 + li) * 32 + half * 16);
    v8f z{};
    v8f s0 = wmma_bf16(qfr, kf0, z);
    v8f s1 = wmma_bf16(qfr, kf1, z);

#pragma unroll
    for (int r = 0; r < 8; ++r) {
      float mx = fmaxf(s0[r], s1[r]);
      mx = fmaxf(mx, __shfl_xor(mx, 1));
      mx = fmaxf(mx, __shfl_xor(mx, 2));
      mx = fmaxf(mx, __shfl_xor(mx, 4));
      mx = fmaxf(mx, __shfl_xor(mx, 8));
      float mnew  = fmaxf(mrun[r], mx);
      float alpha = __expf(mrun[r] - mnew);
      float p0 = __expf(s0[r] - mnew);
      float p1 = __expf(s1[r] - mnew);
      float sm = p0 + p1;
      sm += __shfl_xor(sm, 1);
      sm += __shfl_xor(sm, 2);
      sm += __shfl_xor(sm, 4);
      sm += __shfl_xor(sm, 8);
      lrun[r] = lrun[r] * alpha + sm;
      mrun[r] = mnew;
#pragma unroll
      for (int n = 0; n < 4; ++n) accO[n][r] *= alpha;
      int prow = r + half * 8;                       // D-layout row
      myP[prow * 32 + li]      = f2bf(p0);
      myP[prow * 32 + 16 + li] = f2bf(p1);
    }
    __builtin_amdgcn_wave_barrier();                 // keep LDS store->load order

    v8bf plo = *(const v8bf*)(myP + li * 32 + c0);
    v8bf phi = *(const v8bf*)(myP + li * 32 + c0 + 8);
    v16bf pfr = cat8(plo, phi);
#pragma unroll
    for (int n = 0; n < 4; ++n) {
      const __bf16* vp = Vt + ((size_t)(b * SC + n * 16 + li)) * SEQ + kBase + half * 16;
      v16bf vf = *(const v16bf*)vp;
      accO[n] = wmma_bf16(pfr, vf, accO[n]);
    }
    __builtin_amdgcn_wave_barrier();
  }

#pragma unroll
  for (int r = 0; r < 8; ++r) {
    float inv = 1.0f / lrun[r];
    size_t mrow = (size_t)(mBase + r + half * 8) * SC;
#pragma unroll
    for (int n = 0; n < 4; ++n)
      Obf[mrow + n * 16 + li] = f2bf(accO[n][r] * inv);
  }
}

// ---------------------------------------------------------------------------
// K3: head output projection: headout = O @ ow^T + ob.
// Writes into concat (layout [B][C][S], channel offset head*64) and prev
// (row-major [NTOK][64] fp32, input to the next head).
// ---------------------------------------------------------------------------
__global__ void outproj_kernel(const __bf16* __restrict__ Obf, const __bf16* __restrict__ owb,
                               const float* __restrict__ ob, int head,
                               float* __restrict__ concat, float* __restrict__ prevOut)
{
  int wave = (blockIdx.x * blockDim.x + threadIdx.x) >> 5;   // [0, 512)
  int lane = threadIdx.x & 31, li = lane & 15, half = lane >> 4;
  int mBase = wave * 16;
  int b = mBase >> 12;
  int sBase = mBase & (SEQ - 1);
  const __bf16* W = owb + head * 64 * 64;
  const float* bias = ob + head * 64;

  int row = mBase + li;
  v16bf af[2];
#pragma unroll
  for (int t = 0; t < 2; ++t) {
    v8bf lo = *(const v8bf*)(Obf + (size_t)row * 64 + t * 32 + half * 16);
    v8bf hi = *(const v8bf*)(Obf + (size_t)row * 64 + t * 32 + half * 16 + 8);
    af[t] = cat8(lo, hi);
  }

  for (int nt = 0; nt < 4; ++nt) {
    int nB = nt * 16;
    v16bf b0 = *(const v16bf*)(W + (size_t)(nB + li) * 64 + half * 16);
    v16bf b1 = *(const v16bf*)(W + (size_t)(nB + li) * 64 + 32 + half * 16);
    v8f acc{};
    acc = wmma_bf16(af[0], b0, acc);
    acc = wmma_bf16(af[1], b1, acc);
    float bi = bias[nB + li];
#pragma unroll
    for (int r = 0; r < 8; ++r) acc[r] += bi;
    int n = nB + li;
    int s0 = sBase + half * 8;
    *(v8f*)(concat + ((size_t)(b * CCH + head * SC + n)) * SEQ + s0) = acc;
#pragma unroll
    for (int r = 0; r < 8; ++r)
      prevOut[(size_t)(mBase + r + half * 8) * SC + n] = acc[r];
  }
}

// ---------------------------------------------------------------------------
// BN batch stats (biased, matching jnp.var): one block per channel.
// ---------------------------------------------------------------------------
__global__ void bnstats_kernel(const float* __restrict__ data,
                               float* __restrict__ mean, float* __restrict__ invstd)
{
  int c = blockIdx.x;
  float s = 0.f, ss = 0.f;
  for (int b = 0; b < BB; ++b) {
    const float* p = data + ((size_t)(b * CCH + c)) * SEQ;
    for (int i = threadIdx.x; i < SEQ; i += blockDim.x) {
      float v = p[i]; s += v; ss += v * v;
    }
  }
  __shared__ float sh[2][256];
  sh[0][threadIdx.x] = s; sh[1][threadIdx.x] = ss;
  __syncthreads();
  for (int off = 128; off > 0; off >>= 1) {
    if ((int)threadIdx.x < off) {
      sh[0][threadIdx.x] += sh[0][threadIdx.x + off];
      sh[1][threadIdx.x] += sh[1][threadIdx.x + off];
    }
    __syncthreads();
  }
  if (threadIdx.x == 0) {
    float N = (float)(BB * SEQ);
    float m = sh[0][0] / N;
    float var = sh[1][0] / N - m * m;
    mean[c] = m;
    invstd[c] = rsqrtf(var + 1e-5f);
  }
}

// out1 = x + bn(concat); also emit bf16 token-major copy for the FFN GEMM.
__global__ void bn1_apply_kernel(const float* __restrict__ x, const float* __restrict__ concat,
                                 const float* __restrict__ mean, const float* __restrict__ invstd,
                                 const float* __restrict__ g, const float* __restrict__ bb,
                                 float* __restrict__ out1, __bf16* __restrict__ out1b)
{
  size_t total = (size_t)BB * CCH * SEQ;
  for (size_t idx = (size_t)blockIdx.x * blockDim.x + threadIdx.x; idx < total;
       idx += (size_t)gridDim.x * blockDim.x) {
    size_t s = idx & (SEQ - 1);
    size_t c = (idx >> 12) & (CCH - 1);
    size_t b = idx >> 20;
    float v = x[idx] + (concat[idx] - mean[c]) * invstd[c] * g[c] + bb[c];
    out1[idx] = v;
    out1b[(b * SEQ + s) * CCH + c] = f2bf(v);
  }
}

// ---------------------------------------------------------------------------
// FFN1: H = GELU(out1 @ f1w^T + f1b), [8192 x 1024], K=256.
// One wave per (16-row tile, 128-col group): A frags held in registers,
// 64 WMMAs per wave.
// ---------------------------------------------------------------------------
__global__ void ffn1_kernel(const __bf16* __restrict__ A, const __bf16* __restrict__ W,
                            const float* __restrict__ bias, __bf16* __restrict__ Hout)
{
  int wave = (blockIdx.x * blockDim.x + threadIdx.x) >> 5;   // [0, 4096)
  int lane = threadIdx.x & 31, li = lane & 15, half = lane >> 4;
  int tileM = wave >> 3, ng = wave & 7;
  int mBase = tileM * 16;
  int row = mBase + li;

  v16bf af[8];
#pragma unroll
  for (int t = 0; t < 8; ++t) {
    v8bf lo = *(const v8bf*)(A + (size_t)row * CCH + t * 32 + half * 16);
    v8bf hi = *(const v8bf*)(A + (size_t)row * CCH + t * 32 + half * 16 + 8);
    af[t] = cat8(lo, hi);
  }

  for (int j = 0; j < 8; ++j) {
    int nB = (ng * 8 + j) * 16;
    v8f acc{};
#pragma unroll
    for (int t = 0; t < 8; ++t) {
      v16bf bf = *(const v16bf*)(W + (size_t)(nB + li) * CCH + t * 32 + half * 16);
      acc = wmma_bf16(af[t], bf, acc);
    }
    float bi = bias[nB + li];
#pragma unroll
    for (int r = 0; r < 8; ++r) {
      float v = acc[r] + bi;
      v = 0.5f * v * (1.0f + erff(v * 0.70710678118654752f));   // exact GELU
      Hout[(size_t)(mBase + r + half * 8) * CF + nB + li] = f2bf(v);
    }
  }
}

// FFN2: h2 = H @ f2w^T + f2b, [8192 x 256], K=1024; result in [B][C][S] layout.
__global__ void ffn2_kernel(const __bf16* __restrict__ Hb, const __bf16* __restrict__ W,
                            const float* __restrict__ bias, float* __restrict__ h2)
{
  int wave = (blockIdx.x * blockDim.x + threadIdx.x) >> 5;   // [0, 1024)
  int lane = threadIdx.x & 31, li = lane & 15, half = lane >> 4;
  int tileM = wave >> 1, nh = wave & 1;
  int mBase = tileM * 16;
  int row = mBase + li;
  int b = mBase >> 12;
  int sBase = mBase & (SEQ - 1);

  v8f acc[8];
#pragma unroll
  for (int j = 0; j < 8; ++j) acc[j] = v8f{};

  for (int t = 0; t < 32; ++t) {
    v8bf lo = *(const v8bf*)(Hb + (size_t)row * CF + t * 32 + half * 16);
    v8bf hi = *(const v8bf*)(Hb + (size_t)row * CF + t * 32 + half * 16 + 8);
    v16bf af = cat8(lo, hi);
#pragma unroll
    for (int j = 0; j < 8; ++j) {
      int nRow = (nh * 8 + j) * 16 + li;
      v16bf bf = *(const v16bf*)(W + (size_t)nRow * CF + t * 32 + half * 16);
      acc[j] = wmma_bf16(af, bf, acc[j]);
    }
  }
#pragma unroll
  for (int j = 0; j < 8; ++j) {
    int n = (nh * 8 + j) * 16 + li;
    float bi = bias[n];
#pragma unroll
    for (int r = 0; r < 8; ++r) acc[j][r] += bi;
    int s0 = sBase + half * 8;
    *(v8f*)(h2 + ((size_t)(b * CCH + n)) * SEQ + s0) = acc[j];
  }
}

// final = out1 + bn(h2)
__global__ void final_kernel(const float* __restrict__ out1, const float* __restrict__ h2,
                             const float* __restrict__ mean, const float* __restrict__ invstd,
                             const float* __restrict__ g, const float* __restrict__ bb,
                             float* __restrict__ out)
{
  size_t total = (size_t)BB * CCH * SEQ;
  for (size_t idx = (size_t)blockIdx.x * blockDim.x + threadIdx.x; idx < total;
       idx += (size_t)gridDim.x * blockDim.x) {
    size_t c = (idx >> 12) & (CCH - 1);
    out[idx] = out1[idx] + (h2[idx] - mean[c]) * invstd[c] * g[c] + bb[c];
  }
}

// ---------------------------------------------------------------------------
extern "C" void kernel_launch(void* const* d_in, const int* in_sizes, int n_in,
                              void* d_out, int out_size, void* d_ws, size_t ws_size,
                              hipStream_t stream)
{
  (void)in_sizes; (void)n_in; (void)out_size; (void)ws_size;
  const float* x   = (const float*)d_in[0];
  const float* qw  = (const float*)d_in[1];
  const float* qb  = (const float*)d_in[2];
  const float* kw  = (const float*)d_in[3];
  const float* kb  = (const float*)d_in[4];
  const float* vw  = (const float*)d_in[5];
  const float* vb  = (const float*)d_in[6];
  const float* qlw = (const float*)d_in[7];
  const float* klw = (const float*)d_in[8];
  const float* ow  = (const float*)d_in[9];
  const float* ob  = (const float*)d_in[10];
  const float* f1w = (const float*)d_in[11];
  const float* f1b = (const float*)d_in[12];
  const float* f2w = (const float*)d_in[13];
  const float* f2b = (const float*)d_in[14];
  const float* g1  = (const float*)d_in[15];
  const float* b1  = (const float*)d_in[16];
  const float* g2  = (const float*)d_in[17];
  const float* b2  = (const float*)d_in[18];

  char* ws = (char*)d_ws;
  size_t off = 0;
  auto alloc = [&](size_t bytes) -> char* {
    char* p = ws + off;
    off = (off + bytes + 255) & ~(size_t)255;
    return p;
  };
  __bf16* Wcat  = (__bf16*)alloc((size_t)4 * 96 * 64 * 2);
  float*  bcat  = (float*) alloc((size_t)4 * 96 * 4);
  __bf16* owb   = (__bf16*)alloc((size_t)4 * 64 * 64 * 2);
  __bf16* f1wb  = (__bf16*)alloc((size_t)CF * CCH * 2);
  __bf16* f2wb  = (__bf16*)alloc((size_t)CCH * CF * 2);
  float*  prev  = (float*) alloc((size_t)NTOK * SC * 4);
  __bf16* Vt    = (__bf16*)alloc((size_t)BB * SC * SEQ * 2);
  __bf16* Qlb   = (__bf16*)alloc((size_t)NTOK * 32 * 2);
  __bf16* Klb   = (__bf16*)alloc((size_t)NTOK * 32 * 2);
  __bf16* Obf   = (__bf16*)alloc((size_t)NTOK * SC * 2);
  float*  cc    = (float*) alloc((size_t)BB * CCH * SEQ * 4);
  float*  out1  = (float*) alloc((size_t)BB * CCH * SEQ * 4);
  __bf16* out1b = (__bf16*)alloc((size_t)NTOK * CCH * 2);
  __bf16* Hbf   = (__bf16*)alloc((size_t)NTOK * CF * 2);
  float*  h2    = (float*) alloc((size_t)BB * CCH * SEQ * 4);
  float*  mean1 = (float*) alloc(CCH * 4);
  float*  istd1 = (float*) alloc(CCH * 4);
  float*  mean2 = (float*) alloc(CCH * 4);
  float*  istd2 = (float*) alloc(CCH * 4);

  prep_kernel<<<64, 256, 0, stream>>>(qw, qb, kw, kb, vw, vb, qlw, klw, ow, f1w, f2w,
                                      Wcat, bcat, owb, f1wb, f2wb);
  for (int i = 0; i < 4; ++i) {
    proj_kernel<<<64, 256, 0, stream>>>(x, (i == 0) ? nullptr : prev, Wcat, bcat, i,
                                        Vt, Qlb, Klb);
    attn_kernel<<<64, 256, 0, stream>>>(Qlb, Klb, Vt, Obf);
    outproj_kernel<<<64, 256, 0, stream>>>(Obf, owb, ob, i, cc, prev);
  }
  bnstats_kernel<<<256, 256, 0, stream>>>(cc, mean1, istd1);
  bn1_apply_kernel<<<2048, 256, 0, stream>>>(x, cc, mean1, istd1, g1, b1, out1, out1b);
  ffn1_kernel<<<512, 256, 0, stream>>>(out1b, f1wb, f1b, Hbf);
  ffn2_kernel<<<128, 256, 0, stream>>>(Hbf, f2wb, f2b, h2);
  bnstats_kernel<<<256, 256, 0, stream>>>(h2, mean2, istd2);
  final_kernel<<<2048, 256, 0, stream>>>(out1, h2, mean2, istd2, g2, b2, (float*)d_out);
}
